// RingAttention_50027779064326
// MI455X (gfx1250) — matrix operand order repaired and checked by reading, not compiled, not run on hardware
//
#include <hip/hip_runtime.h>

// ---------------------------------------------------------------------------
// Causal attention block for MI455X (gfx1250, wave32, WMMA bf16 16x16x32).
// Pipeline: f32->bf16 convert | QKV GEMM | flash attention | out-proj GEMM.
// Data movement: Tensor Data Mover (tensor_load_to_lds) for strided 2D tiles
// with hardware LDS row padding; manual staging only where a transpose is
// required (B / V tiles).
// ---------------------------------------------------------------------------

typedef __attribute__((ext_vector_type(16))) __bf16 v16bf;
typedef __attribute__((ext_vector_type(8)))  float  v8f;
typedef __attribute__((ext_vector_type(4)))  unsigned int u32x4;
typedef __attribute__((ext_vector_type(4)))  int  i32x4;
typedef __attribute__((ext_vector_type(8)))  int  i32x8;

#define SCALE_Q 0.125f   // 64^-0.5

__device__ __forceinline__ __bf16 f2bf(float x) {
  unsigned u = __builtin_bit_cast(unsigned, x);
  unsigned r = u + 0x7FFFu + ((u >> 16) & 1u);   // round-to-nearest-even
  unsigned short h = (unsigned short)(r >> 16);
  return __builtin_bit_cast(__bf16, h);
}

union Frag32 { uint4 q[2]; v16bf v; };
union Vec8   { uint4 q; __bf16 e[8]; };

// A-matrix fragment (16xK tile, K-chunk of 32). Lane l<16: row=l, K {0..7,16..23};
// lane l+16: row=l, K {8..15,24..31}.  base points at (row0, k0) of the tile.
__device__ __forceinline__ v16bf load_fragA(const __bf16* base, int stride, int kbase) {
  const int lane = threadIdx.x & 31;
  const __bf16* p = base + (lane & 15) * stride + kbase + ((lane & 16) ? 8 : 0);
  Frag32 f;
  f.q[0] = *(const uint4*)(p);
  f.q[1] = *(const uint4*)(p + 16);
  return f.v;
}

// B-matrix fragment from [N][K]-transposed storage. Lane l<16: col=l, K {0..7,16..23};
// lane l+16: col=l, K {8..15,24..31}.
__device__ __forceinline__ v16bf load_fragB(const __bf16* base, int stride, int kbase) {
  const int lane = threadIdx.x & 31;
  const __bf16* p = base + (lane & 15) * stride + kbase + ((lane & 16) ? 8 : 0);
  Frag32 f;
  f.q[0] = *(const uint4*)(p);
  f.q[1] = *(const uint4*)(p + 16);
  return f.v;
}

__device__ __forceinline__ v8f wmma_bf16(v16bf a, v16bf b, v8f c) {
  return __builtin_amdgcn_wmma_f32_16x16x32_bf16(false, a, false, b, (short)0, c,
                                                 false, false);
}

// ---------------------------------------------------------------------------
// Tensor Data Mover: 2D tile of 2-byte elements, global (row stride in
// elements) -> LDS, with optional per-row LDS padding.
//   pad_interval field: rows of (1<<(pi+1)) DWORDs;  pad_amount field: pa+1 DWORDs.
// Issue from ONE wave per block; follow with s_wait_tensorcnt + barrier.
// NOTE: this toolchain exposes the 6-operand builtin
//   (u32x4 g0, i32x8 g1, i32x4 g2, i32x4 g3, i32x8 aux, i32 cpol).
__device__ __forceinline__ void tdm_load_2d_bf16(
    const void* gptr, const void* lds_ptr,
    unsigned tile_d0, unsigned tile_d1,
    unsigned long long stride_elems,
    unsigned pad_interval_f, unsigned pad_amount_f) {
  unsigned long long ga = (unsigned long long)gptr;
  unsigned lds_addr = (unsigned)(unsigned long long)lds_ptr;   // low 32 = LDS offset

  u32x4 g0 = { 1u,                                   // count=1, user desc
               lds_addr,
               (unsigned)ga,
               (unsigned)(ga >> 32) | (2u << 30) };  // global_addr hi, type=2

  unsigned td0 = (stride_elems > 0xFFFFFFFFull) ? 0xFFFFFFFFu
                                                : (unsigned)stride_elems;
  unsigned td1 = 0x100000u;                          // generous dim1 bound
  unsigned w0 = (1u << 16)                           // data_size = 2 bytes
              | (1u << 20)                           // pad_enable
              | (pad_interval_f << 22) | (pad_amount_f << 25);
  i32x8 g1 = { (int)w0,
               (int)((td0 & 0xFFFFu) << 16),         // atomic_addr=0 | td0.lo
               (int)((td0 >> 16) | ((td1 & 0xFFFFu) << 16)),
               (int)(((td1 >> 16) & 0xFFFFu) | (tile_d0 << 16)),
               (int)(tile_d1 & 0xFFFFu),             // tile_dim1 | tile_dim2=0
               (int)(unsigned)stride_elems,          // dim0_stride lo32
               (int)((unsigned)(stride_elems >> 32) & 0xFFFFu),
               0 };
  i32x4 g2 = { 0, 0, 0, 0 };
  i32x4 g3 = { 0, 0, 0, 0 };
  i32x8 g4 = { 0, 0, 0, 0, 0, 0, 0, 0 };
  __builtin_amdgcn_tensor_load_to_lds(g0, g1, g2, g3, g4, 0);
}

// ---------------------------------------------------------------------------
__global__ void k_f32_to_bf16(const float* __restrict__ src,
                              __bf16* __restrict__ dst, int n) {
  int i = blockIdx.x * blockDim.x + threadIdx.x;
  int stride = gridDim.x * blockDim.x;
  for (; i < n; i += stride) dst[i] = f2bf(src[i]);
}

// ---------------------------------------------------------------------------
// 128x128 block tile GEMM core: 256 threads = 8 waves (4 M x 2 N), each wave
// computes 32x64 via 2x4 WMMA accumulators. K staged 32 at a time:
//   A tile via TDM (wave 0) with hardware row padding (stride 40 bf16),
//   B tile manually transposed to [N][K] while the DMA runs.
struct Acc8 { v8f c[2][4]; };

__device__ __forceinline__ void gemm_core_128x128(
    const __bf16* __restrict__ A, int lda,
    const __bf16* __restrict__ Bm, int ldb,
    int K, int rowBase, int colBase,
    __bf16* As, __bf16* BsT, Acc8& acc) {
  const int tid = threadIdx.x;
  const int wid = tid >> 5;
  const int waveM = wid >> 1;
  const int waveN = wid & 1;

  v8f z;
#pragma unroll
  for (int i = 0; i < 8; ++i) z[i] = 0.f;
#pragma unroll
  for (int mt = 0; mt < 2; ++mt)
#pragma unroll
    for (int nt = 0; nt < 4; ++nt) acc.c[mt][nt] = z;

  for (int kk = 0; kk < K; kk += 32) {
    __syncthreads();
    // A tile: 128 rows x 32 K -> LDS stride 40 (rows of 16 DWORDs + 4 DWORD pad).
    if (wid == 0)
      tdm_load_2d_bf16(&A[(size_t)rowBase * lda + kk], As,
                       /*tile_d0=*/32, /*tile_d1=*/128,
                       (unsigned long long)lda,
                       /*pad_interval(16dw)=*/3, /*pad_amount(4dw)=*/3);
    // B tile: 32 K x 128 N, transposed to [N][K] while the DMA runs.
#pragma unroll
    for (int i = 0; i < 2; ++i) {
      int idx = tid + i * 256;
      int k = idx >> 4, cc = idx & 15;
      Vec8 t;
      t.q = *(const uint4*)&Bm[(size_t)(kk + k) * ldb + colBase + cc * 8];
#pragma unroll
      for (int e = 0; e < 8; ++e) BsT[(cc * 8 + e) * 40 + k] = t.e[e];
    }
    if (wid == 0) __builtin_amdgcn_s_wait_tensorcnt(0);
    __syncthreads();

    v16bf af[2];
#pragma unroll
    for (int mt = 0; mt < 2; ++mt)
      af[mt] = load_fragA(As + (waveM * 32 + mt * 16) * 40, 40, 0);
#pragma unroll
    for (int nt = 0; nt < 4; ++nt) {
      v16bf bf = load_fragB(BsT + (waveN * 64 + nt * 16) * 40, 40, 0);
#pragma unroll
      for (int mt = 0; mt < 2; ++mt)
        acc.c[mt][nt] = wmma_bf16(af[mt], bf, acc.c[mt][nt]);
    }
  }
}

// ---------------------------------------------------------------------------
// QKV projection: [8192,1024] x [1024,3072] -> scatter into per-head q/k/v
// [b][h][n][64] bf16, softmax scale folded into Q. Segment (q/k/v) is
// block-uniform, so pointer + scale are selected once (no per-element branches).
__global__ void __launch_bounds__(256) k_qkv_gemm(
    const __bf16* __restrict__ xb, const __bf16* __restrict__ wb,
    __bf16* __restrict__ qv, __bf16* __restrict__ kv, __bf16* __restrict__ vv) {
  __shared__ __align__(16) __bf16 As[128 * 40];
  __shared__ __align__(16) __bf16 BsT[128 * 40];
  Acc8 acc;
  const int rowBase = blockIdx.x * 128, colBase = blockIdx.y * 128;
  gemm_core_128x128(xb, 1024, wb, 3072, 1024, rowBase, colBase, As, BsT, acc);

  const int seg = colBase >> 10;                 // 0=q, 1=k, 2=v (uniform)
  __bf16* dst = (seg == 0) ? qv : (seg == 1) ? kv : vv;
  const float scl = (seg == 0) ? SCALE_Q : 1.0f;
  const int colSeg = colBase & 1023;

  const int lane = threadIdx.x & 31, hvn = lane >> 4, ln = lane & 15;
  const int wid = threadIdx.x >> 5, waveM = wid >> 1, waveN = wid & 1;
#pragma unroll
  for (int mt = 0; mt < 2; ++mt)
#pragma unroll
    for (int nt = 0; nt < 4; ++nt)
#pragma unroll
      for (int j = 0; j < 8; ++j) {
        int mrow = rowBase + waveM * 32 + mt * 16 + j + 8 * hvn;
        int col = colSeg + waveN * 64 + nt * 16 + ln;
        int h = col >> 6, d = col & 63;
        int bb = mrow >> 12, n = mrow & 4095;
        dst[(((size_t)bb * 16 + h) * 4096 + n) * 64 + d] =
            f2bf(acc.c[mt][nt][j] * scl);
      }
}

// ---------------------------------------------------------------------------
// Flash attention: one block per (b, h, 64-row q block); 4 waves x 16 q rows.
// Causal tile skipping; online softmax; QK^T and PV via bf16 WMMA.
// K tile staged by TDM (natural layout, padded rows); V staged transposed.
__global__ void __launch_bounds__(128) k_attention(
    const __bf16* __restrict__ q, const __bf16* __restrict__ k,
    const __bf16* __restrict__ v, __bf16* __restrict__ attn) {
  __shared__ __align__(16) __bf16 Ks[64 * 72];        // [key][d]  (B for QK^T)
  __shared__ __align__(16) __bf16 VsT[64 * 72];       // [d][key]  (B for PV)
  __shared__ __align__(16) __bf16 Ps[4 * 16 * 72];    // per-wave P (A for PV)

  const int qb = blockIdx.x, h = blockIdx.y, b = blockIdx.z;
  const int tid = threadIdx.x, wid = tid >> 5, lane = tid & 31;
  const int hvn = lane >> 4, ln = lane & 15;

  const size_t headOff = (((size_t)b * 16 + h) * 4096) * 64;
  const __bf16* qh = q + headOff;
  const __bf16* kh = k + headOff;
  const __bf16* vh = v + headOff;
  const int qrow0 = qb * 64 + wid * 16;

  v16bf aq[2];
#pragma unroll
  for (int c = 0; c < 2; ++c)
    aq[c] = load_fragA(qh + (size_t)qrow0 * 64, 64, c * 32);

  float mrow[8], lrow[8];
  v8f o[4];
#pragma unroll
  for (int j = 0; j < 8; ++j) { mrow[j] = -1e30f; lrow[j] = 0.f; }
#pragma unroll
  for (int dt = 0; dt < 4; ++dt)
#pragma unroll
    for (int j = 0; j < 8; ++j) o[dt][j] = 0.f;

  __bf16* Pw = Ps + wid * 16 * 72;

  for (int jt = 0; jt <= qb; ++jt) {
    __syncthreads();
    // K tile: 64x64 -> LDS stride 72 via TDM (rows of 32 DWORDs + 4 DWORD pad).
    if (wid == 0)
      tdm_load_2d_bf16(&kh[(size_t)jt * 64 * 64], Ks,
                       /*tile_d0=*/64, /*tile_d1=*/64,
                       /*stride=*/64ull,
                       /*pad_interval(32dw)=*/4, /*pad_amount(4dw)=*/3);
    // V tile transposed to [d][key] while the DMA runs.
#pragma unroll
    for (int i = 0; i < 4; ++i) {
      int idx = tid + i * 128;
      int r = idx >> 3, c8 = idx & 7;
      Vec8 t;
      t.q = *(const uint4*)&vh[((size_t)jt * 64 + r) * 64 + c8 * 8];
#pragma unroll
      for (int e = 0; e < 8; ++e) VsT[(c8 * 8 + e) * 72 + r] = t.e[e];
    }
    // Prefetch next iteration's K/V tiles into cache.
    if (jt < qb) {
      __builtin_prefetch(&kh[(size_t)(jt + 1) * 64 * 64], 0, 1);
      __builtin_prefetch(&vh[(size_t)(jt + 1) * 64 * 64], 0, 1);
    }
    if (wid == 0) __builtin_amdgcn_s_wait_tensorcnt(0);
    __syncthreads();

    // S = Q K^T  (16 rows x 64 keys per wave)
    v8f s[4];
#pragma unroll
    for (int nt = 0; nt < 4; ++nt)
#pragma unroll
      for (int j = 0; j < 8; ++j) s[nt][j] = 0.f;
#pragma unroll
    for (int c = 0; c < 2; ++c)
#pragma unroll
      for (int nt = 0; nt < 4; ++nt) {
        v16bf bk = load_fragB(Ks + nt * 16 * 72, 72, c * 32);
        s[nt] = wmma_bf16(aq[c], bk, s[nt]);
      }

    // Causal mask on the diagonal tile (scale was folded into Q already).
    if (jt == qb) {
#pragma unroll
      for (int nt = 0; nt < 4; ++nt)
#pragma unroll
        for (int j = 0; j < 8; ++j) {
          int colg = jt * 64 + nt * 16 + ln;
          int rowg = qrow0 + j + 8 * hvn;
          if (colg > rowg) s[nt][j] = -1e30f;
        }
    }

    // Online softmax: row stats via xor-shuffles within 16-lane halves.
#pragma unroll
    for (int j = 0; j < 8; ++j) {
      float mn = s[0][j];
#pragma unroll
      for (int nt = 1; nt < 4; ++nt) mn = fmaxf(mn, s[nt][j]);
#pragma unroll
      for (int msk = 8; msk >= 1; msk >>= 1)
        mn = fmaxf(mn, __shfl_xor(mn, msk, 32));
      float mtot = fmaxf(mrow[j], mn);
      float corr = __expf(mrow[j] - mtot);
      mrow[j] = mtot;
      float rs = 0.f;
#pragma unroll
      for (int nt = 0; nt < 4; ++nt) {
        float p = __expf(s[nt][j] - mtot);
        s[nt][j] = p;
        rs += p;
      }
#pragma unroll
      for (int msk = 8; msk >= 1; msk >>= 1)
        rs += __shfl_xor(rs, msk, 32);
      lrow[j] = lrow[j] * corr + rs;
#pragma unroll
      for (int dt = 0; dt < 4; ++dt) o[dt][j] *= corr;
    }

    // C-layout f32 P -> bf16 A-layout via per-wave LDS round-trip.
#pragma unroll
    for (int nt = 0; nt < 4; ++nt)
#pragma unroll
      for (int j = 0; j < 8; ++j)
        Pw[(j + 8 * hvn) * 72 + nt * 16 + ln] = f2bf(s[nt][j]);

    // O += P V
#pragma unroll
    for (int c = 0; c < 2; ++c) {
      v16bf ap = load_fragA(Pw, 72, c * 32);
#pragma unroll
      for (int dt = 0; dt < 4; ++dt) {
        v16bf bv = load_fragB(VsT + dt * 16 * 72, 72, c * 32);
        o[dt] = wmma_bf16(ap, bv, o[dt]);
      }
    }
  }

  // Normalize and write attention output [b][n][h*64+d] as bf16.
#pragma unroll
  for (int j = 0; j < 8; ++j) {
    float inv = 1.0f / lrow[j];
    int rowg = qrow0 + j + 8 * hvn;
#pragma unroll
    for (int dt = 0; dt < 4; ++dt) {
      int d = dt * 16 + ln;
      attn[((size_t)b * 4096 + rowg) * 1024 + h * 64 + d] = f2bf(o[dt][j] * inv);
    }
  }
}

// ---------------------------------------------------------------------------
// Output projection: [8192,1024] x [1024,1024] -> fp32 d_out.
__global__ void __launch_bounds__(256) k_out_gemm(
    const __bf16* __restrict__ ab, const __bf16* __restrict__ wb,
    float* __restrict__ out) {
  __shared__ __align__(16) __bf16 As[128 * 40];
  __shared__ __align__(16) __bf16 BsT[128 * 40];
  Acc8 acc;
  const int rowBase = blockIdx.x * 128, colBase = blockIdx.y * 128;
  gemm_core_128x128(ab, 1024, wb, 1024, 1024, rowBase, colBase, As, BsT, acc);

  const int lane = threadIdx.x & 31, hvn = lane >> 4, ln = lane & 15;
  const int wid = threadIdx.x >> 5, waveM = wid >> 1, waveN = wid & 1;
#pragma unroll
  for (int mt = 0; mt < 2; ++mt)
#pragma unroll
    for (int nt = 0; nt < 4; ++nt)
#pragma unroll
      for (int j = 0; j < 8; ++j) {
        int mrow = rowBase + waveM * 32 + mt * 16 + j + 8 * hvn;
        int col = colBase + waveN * 64 + nt * 16 + ln;
        out[(size_t)mrow * 1024 + col] = acc.c[mt][nt][j];
      }
}

// ---------------------------------------------------------------------------
extern "C" void kernel_launch(void* const* d_in, const int* in_sizes, int n_in,
                              void* d_out, int out_size, void* d_ws, size_t ws_size,
                              hipStream_t stream) {
  (void)in_sizes; (void)n_in; (void)out_size; (void)ws_size;
  const float* x     = (const float*)d_in[0];
  const float* w_qkv = (const float*)d_in[1];
  const float* w_out = (const float*)d_in[2];
  float* out = (float*)d_out;

  // Workspace layout (bf16 elements): ~92 MB total.
  __bf16* p = (__bf16*)d_ws;
  const size_t NX = (size_t)2 * 4096 * 1024;          // 8,388,608
  __bf16* xb    = p; p += NX;
  __bf16* wqkvb = p; p += (size_t)1024 * 3072;
  __bf16* woutb = p; p += (size_t)1024 * 1024;
  __bf16* qb    = p; p += NX;
  __bf16* kb    = p; p += NX;
  __bf16* vb    = p; p += NX;
  __bf16* attn  = p; p += NX;

  k_f32_to_bf16<<<2048, 256, 0, stream>>>(x, xb, (int)NX);
  k_f32_to_bf16<<<2048, 256, 0, stream>>>(w_qkv, wqkvb, 1024 * 3072);
  k_f32_to_bf16<<<1024, 256, 0, stream>>>(w_out, woutb, 1024 * 1024);

  k_qkv_gemm<<<dim3(64, 24), 256, 0, stream>>>(xb, wqkvb, qb, kb, vb);
  k_attention<<<dim3(64, 16, 2), 128, 0, stream>>>(qb, kb, vb, attn);
  k_out_gemm<<<dim3(64, 8), 256, 0, stream>>>(attn, woutb, out);
}
